// MllamaTextSdpaAttention_72318659330550
// MI455X (gfx1250) — matrix operand order repaired and checked by reading, not compiled, not run on hardware
//
#include <hip/hip_runtime.h>
#include <hip/hip_bf16.h>

// MI455X (gfx1250) GQA attention block.
// B=2, S=1024, DIM=4096, H=32, KV=8, HD=128. bf16 WMMA (16x16x32) everywhere,
// f32 accumulate. Memory-bound op (~250MB traffic, ~206 GFLOP) -> bf16 halves
// intermediate traffic; GEMMs double-buffer LDS with gfx1250 async global->LDS
// (ASYNCcnt) and fragment loads are batched so one s_wait_dscnt covers a group
// of WMMAs instead of one wait per WMMA.

#define SEQ   1024
#define BATCH 2
#define DIM   4096
#define NH    32
#define NKV   8
#define HD    128

typedef __attribute__((ext_vector_type(16))) __bf16 v16bf;
typedef __attribute__((ext_vector_type(8)))  float  v8f;
typedef __attribute__((ext_vector_type(4)))  int    v4i;

union Frag16 { uint4 u[2]; v16bf v; };

#if defined(__has_builtin)
#if __has_builtin(__builtin_amdgcn_global_load_async_to_lds_b128) && \
    __has_builtin(__builtin_amdgcn_s_wait_asynccnt)
#define HAVE_ASYNC_LDS 1
#endif
#endif
#ifndef HAVE_ASYNC_LDS
#define HAVE_ASYNC_LDS 0
#endif

#if HAVE_ASYNC_LDS
#define LDS_V4I(p) ((__attribute__((address_space(3))) v4i*)(p))
#define GLB_V4I(p) ((v4i*)(p))
#endif

__device__ __forceinline__ unsigned short f2bf(float f) {
    unsigned int u = __float_as_uint(f);
    u += 0x7FFFu + ((u >> 16) & 1u);           // round-to-nearest-even
    return (unsigned short)(u >> 16);
}
__device__ __forceinline__ float bf2f(unsigned short h) {
    return __uint_as_float(((unsigned int)h) << 16);
}
__device__ __forceinline__ v8f zero_v8f() {
    v8f z;
#pragma unroll
    for (int i = 0; i < 8; ++i) z[i] = 0.f;
    return z;
}

// ---------------------------------------------------------------- converts --
__global__ __launch_bounds__(256) void f32_to_bf16_kernel(
    const float* __restrict__ x, unsigned short* __restrict__ y, long n) {
    long i = (long)blockIdx.x * 256 + threadIdx.x;
    if (i < n) y[i] = f2bf(x[i]);
}

// W[K][N] f32 -> Wt[N][K] bf16 (so GEMM B-fragments read contiguous K)
__global__ __launch_bounds__(256) void transpose_to_bf16_kernel(
    const float* __restrict__ W, unsigned short* __restrict__ Wt, int K, int N) {
    __shared__ float tile[32][33];
    int n0 = blockIdx.x * 32, k0 = blockIdx.y * 32;
    for (int i = threadIdx.y; i < 32; i += 8)
        tile[i][threadIdx.x] = W[(size_t)(k0 + i) * N + n0 + threadIdx.x];
    __syncthreads();
    for (int i = threadIdx.y; i < 32; i += 8)
        Wt[(size_t)(n0 + i) * K + k0 + threadIdx.x] = f2bf(tile[threadIdx.x][i]);
}

// ---------------------------------------------------------------- GEMM ------
// C[M,N] = A[M,K] * B[K,N], A bf16 row-major, Bt = B^T bf16 (N x K).
// Block: 256 thr = 8 waves (2x4), wave does 32x32, block tile 64x128, BK=32.
// Double-buffered LDS; async global->LDS (ASYNCcnt) when available.
#define GBM 64
#define GBN 128
#define GBK 32
#define SAS 40   // sA row stride (elems), 80B: keeps 16B alignment, skews banks
#define SBS 40

__device__ __forceinline__ void gemm_tile_compute(
    const unsigned short* __restrict__ sA, const unsigned short* __restrict__ sB,
    v8f acc[2][2], int wm, int wn, int hlf, int l16) {
    Frag16 af[2], bf[2];
#pragma unroll
    for (int mi = 0; mi < 2; ++mi) {
        const int arow = (wm * 32 + mi * 16 + l16) * SAS;
        af[mi].u[0] = *(const uint4*)&sA[arow + hlf * 8];
        af[mi].u[1] = *(const uint4*)&sA[arow + 16 + hlf * 8];
    }
#pragma unroll
    for (int ni = 0; ni < 2; ++ni) {
        const int brow = (wn * 32 + ni * 16 + l16) * SBS + hlf * 8;
        bf[ni].u[0] = *(const uint4*)&sB[brow];
        bf[ni].u[1] = *(const uint4*)&sB[brow + 16];
    }
#pragma unroll
    for (int mi = 0; mi < 2; ++mi)
#pragma unroll
        for (int ni = 0; ni < 2; ++ni)
            acc[mi][ni] = __builtin_amdgcn_wmma_f32_16x16x32_bf16(
                false, af[mi].v, false, bf[ni].v, (short)0, acc[mi][ni], false, false);
}

template <int BF16OUT>
__global__ __launch_bounds__(256) void gemm_bf16_kernel(
    const unsigned short* __restrict__ A, const unsigned short* __restrict__ Bt,
    void* __restrict__ Cout, int M, int N, int K) {
    __shared__ alignas(16) unsigned short sA[2][GBM * SAS];
    __shared__ alignas(16) unsigned short sB[2][GBN * SBS];

    const int tid  = threadIdx.x;
    const int wave = tid >> 5, lane = tid & 31;
    const int hlf  = lane >> 4, l16 = lane & 15;
    const int wm   = wave >> 2, wn = wave & 3;      // 2 x 4 wave grid

    const int ar = tid >> 2, ac = (tid & 3) << 3;   // one uint4 / thread / tile
    const unsigned short* Ag  = A  + (size_t)(blockIdx.y * GBM + ar) * K + ac;
    const unsigned short* Bg0 = Bt + (size_t)(blockIdx.x * GBN + ar) * K + ac;
    const unsigned short* Bg1 = Bt + (size_t)(blockIdx.x * GBN + ar + 64) * K + ac;
    const int la  = ar * SAS + ac;
    const int lb0 = ar * SBS + ac;
    const int lb1 = (ar + 64) * SBS + ac;

    v8f acc[2][2];
#pragma unroll
    for (int mi = 0; mi < 2; ++mi)
#pragma unroll
        for (int ni = 0; ni < 2; ++ni) acc[mi][ni] = zero_v8f();

    int buf = 0;
#if HAVE_ASYNC_LDS
    __builtin_amdgcn_global_load_async_to_lds_b128(GLB_V4I(Ag),  LDS_V4I(&sA[0][la]),  0, 0);
    __builtin_amdgcn_global_load_async_to_lds_b128(GLB_V4I(Bg0), LDS_V4I(&sB[0][lb0]), 0, 0);
    __builtin_amdgcn_global_load_async_to_lds_b128(GLB_V4I(Bg1), LDS_V4I(&sB[0][lb1]), 0, 0);
    for (int kt = 0; kt < K; kt += GBK) {
        __builtin_amdgcn_s_wait_asynccnt(0);
        __syncthreads();
        if (kt + GBK < K) {
            const int nb = buf ^ 1, ko = kt + GBK;
            __builtin_amdgcn_global_load_async_to_lds_b128(GLB_V4I(Ag + ko),  LDS_V4I(&sA[nb][la]),  0, 0);
            __builtin_amdgcn_global_load_async_to_lds_b128(GLB_V4I(Bg0 + ko), LDS_V4I(&sB[nb][lb0]), 0, 0);
            __builtin_amdgcn_global_load_async_to_lds_b128(GLB_V4I(Bg1 + ko), LDS_V4I(&sB[nb][lb1]), 0, 0);
        }
        gemm_tile_compute(sA[buf], sB[buf], acc, wm, wn, hlf, l16);
        buf ^= 1;
    }
#else
    uint4 ra  = *(const uint4*)Ag;
    uint4 rb0 = *(const uint4*)Bg0;
    uint4 rb1 = *(const uint4*)Bg1;
    for (int kt = 0; kt < K; kt += GBK) {
        *(uint4*)&sA[buf][la]  = ra;
        *(uint4*)&sB[buf][lb0] = rb0;
        *(uint4*)&sB[buf][lb1] = rb1;
        __syncthreads();
        if (kt + GBK < K) {                 // prefetch next tile into registers
            ra  = *(const uint4*)(Ag  + kt + GBK);
            rb0 = *(const uint4*)(Bg0 + kt + GBK);
            rb1 = *(const uint4*)(Bg1 + kt + GBK);
        }
        gemm_tile_compute(sA[buf], sB[buf], acc, wm, wn, hlf, l16);
        buf ^= 1;
    }
#endif
#pragma unroll
    for (int mi = 0; mi < 2; ++mi)
#pragma unroll
        for (int ni = 0; ni < 2; ++ni)
#pragma unroll
            for (int v = 0; v < 8; ++v) {
                const size_t r = blockIdx.y * GBM + wm * 32 + mi * 16 + hlf * 8 + v;
                const size_t c = blockIdx.x * GBN + wn * 32 + ni * 16 + l16;
                if (BF16OUT)
                    ((unsigned short*)Cout)[r * N + c] = f2bf(acc[mi][ni][v]);
                else
                    ((float*)Cout)[r * N + c] = acc[mi][ni][v];
            }
}

// ---------------------------------------------------------------- RoPE ------
// in  : [B, S, H*128] bf16 (GEMM output)     out : [B, H, S, 128] bf16
__global__ __launch_bounds__(256) void rope_permute_kernel(
    const unsigned short* __restrict__ src, const float* __restrict__ cosb,
    const float* __restrict__ sinb, unsigned short* __restrict__ dst, int H, long n) {
    long i = (long)blockIdx.x * 256 + threadIdx.x;
    if (i >= n) return;
    const int  d2 = (int)(i & 63);
    long t = i >> 6;
    const int  h  = (int)(t % H);  t /= H;
    const int  s  = (int)(t % SEQ);
    const int  b  = (int)(t / SEQ);
    const float c  = cosb[((size_t)b * SEQ + s) * 64 + d2];
    const float sn = sinb[((size_t)b * SEQ + s) * 64 + d2];
    const unsigned short* p = src + (((size_t)b * SEQ + s) * H + h) * HD + 2 * d2;
    const float xe = bf2f(p[0]), xo = bf2f(p[1]);
    unsigned short* q = dst + (((size_t)b * H + h) * SEQ + s) * HD + 2 * d2;
    q[0] = f2bf(xe * c - xo * sn);
    q[1] = f2bf(xe * sn + xo * c);
}

// [B, S, H*128] -> [B, H, S, 128]  (bf16 copy, for V)
__global__ __launch_bounds__(256) void permute_heads_kernel(
    const unsigned short* __restrict__ src, unsigned short* __restrict__ dst,
    int H, long n) {
    long i = (long)blockIdx.x * 256 + threadIdx.x;
    if (i >= n) return;
    const int d = (int)(i & (HD - 1));
    long t = i >> 7;
    const int h = (int)(t % H);  t /= H;
    const int s = (int)(t % SEQ);
    const int b = (int)(t / SEQ);
    dst[(((size_t)b * H + h) * SEQ + s) * HD + d] =
        src[(((size_t)b * SEQ + s) * H + h) * HD + d];
}

// ---------------------------------------------------------------- attention -
// Flash attention, causal. Block = (qtile of 64 rows, head, batch), 4 waves,
// each wave owns 16 q rows. QK^T and PV both on v_wmma_f32_16x16x32_bf16.
// Q fragments hoisted out of the key loop; B fragments batched so one dscnt
// wait covers 4 WMMAs.
#define QSTR 136   // 64x128 tile row stride (elems); 272B, 16B aligned
#define VSTR 72    // V^T / P row stride; 144B, 16B aligned

__global__ __launch_bounds__(128) void attention_kernel(
    const unsigned short* __restrict__ Q,   // [B, 32, S, 128]
    const unsigned short* __restrict__ Kr,  // [B, 8, S, 128]
    const unsigned short* __restrict__ V,   // [B, 8, S, 128]
    unsigned short* __restrict__ O) {       // [B, S, 4096]
    __shared__ alignas(16) unsigned short sQ[64 * QSTR];
    __shared__ alignas(16) unsigned short sK[64 * QSTR];
    __shared__ alignas(16) unsigned short sVt[HD * VSTR];      // transposed V
    __shared__ alignas(16) unsigned short sP[4][16 * VSTR];    // per-wave P

    const int qt = blockIdx.x, h = blockIdx.y, b = blockIdx.z;
    const int g  = h >> 2;                                     // kv head
    const int tid = threadIdx.x, wave = tid >> 5, lane = tid & 31;
    const int hlf = lane >> 4, l16 = lane & 15;
    const int qbase = qt * 64;
    const float scale = 0.08838834764831845f;                  // 1/sqrt(128)

    const unsigned short* Qg = Q + (((size_t)b * NH + h) * SEQ + qbase) * HD;
    for (int i = tid; i < 64 * 16; i += 128) {
        const int r = i >> 4, c = (i & 15) << 3;
        *(uint4*)&sQ[r * QSTR + c] = *(const uint4*)(Qg + (size_t)r * HD + c);
    }
    __syncthreads();

    // Q fragments for this wave's 16 rows, 4 K-chunks of 32 (loop invariant)
    Frag16 af[4];
    {
        const int qrow = (wave * 16 + l16) * QSTR;
#pragma unroll
        for (int kc = 0; kc < 4; ++kc) {
            af[kc].u[0] = *(const uint4*)&sQ[qrow + kc * 32 + hlf * 8];
            af[kc].u[1] = *(const uint4*)&sQ[qrow + kc * 32 + 16 + hlf * 8];
        }
    }

    v8f oacc[8];
#pragma unroll
    for (int dt = 0; dt < 8; ++dt) oacc[dt] = zero_v8f();
    float rmax[8], rsum[8];
#pragma unroll
    for (int v = 0; v < 8; ++v) { rmax[v] = -1e30f; rsum[v] = 0.f; }

    for (int kt = 0; kt <= qt; ++kt) {                          // causal skip
        const int kbase = kt * 64;
        const unsigned short* Kg = Kr + (((size_t)b * NKV + g) * SEQ + kbase) * HD;
        const unsigned short* Vg = V  + (((size_t)b * NKV + g) * SEQ + kbase) * HD;
        __syncthreads();
        for (int i = tid; i < 64 * 16; i += 128) {
            const int r = i >> 4, c = (i & 15) << 3;
            *(uint4*)&sK[r * QSTR + c] = *(const uint4*)(Kg + (size_t)r * HD + c);
            uint4 vv = *(const uint4*)(Vg + (size_t)r * HD + c);
            const unsigned short* pv = (const unsigned short*)&vv;
#pragma unroll
            for (int e = 0; e < 8; ++e) sVt[(c + e) * VSTR + r] = pv[e];
        }
        __syncthreads();

        // scores: 16 q rows x 64 keys = 4 fragments; batch 8 ds_loads / 4 wmma
        v8f sf[4];
#pragma unroll
        for (int nt = 0; nt < 4; ++nt) {
            Frag16 bf[4];
#pragma unroll
            for (int kc = 0; kc < 4; ++kc) {
                const int brow = (nt * 16 + l16) * QSTR + kc * 32 + hlf * 8;
                bf[kc].u[0] = *(const uint4*)&sK[brow];
                bf[kc].u[1] = *(const uint4*)&sK[brow + 16];
            }
            v8f s = zero_v8f();
#pragma unroll
            for (int kc = 0; kc < 4; ++kc)
                s = __builtin_amdgcn_wmma_f32_16x16x32_bf16(
                    false, af[kc].v, false, bf[kc].v, (short)0, s, false, false);
            sf[nt] = s;
        }
        // scale + causal mask + row max (rows live per-VGPR in lane halves)
        float nmax[8];
#pragma unroll
        for (int v = 0; v < 8; ++v) {
            const int qg = qbase + wave * 16 + hlf * 8 + v;
            float m = rmax[v];
#pragma unroll
            for (int nt = 0; nt < 4; ++nt) {
                const int kg = kbase + nt * 16 + l16;
                float x = sf[nt][v] * scale + (kg <= qg ? 0.f : -1e9f);
                sf[nt][v] = x;
                m = fmaxf(m, x);
            }
            nmax[v] = m;
        }
#pragma unroll
        for (int off = 1; off < 16; off <<= 1)
#pragma unroll
            for (int v = 0; v < 8; ++v)
                nmax[v] = fmaxf(nmax[v], __shfl_xor(nmax[v], off, 32));
        float lsum[8];
#pragma unroll
        for (int v = 0; v < 8; ++v) {
            const float alpha = __expf(rmax[v] - nmax[v]);
            rmax[v] = nmax[v];
            rsum[v] *= alpha;
            lsum[v] = 0.f;
#pragma unroll
            for (int dt = 0; dt < 8; ++dt) oacc[dt][v] *= alpha;
        }
        // P = exp(s - max), stash to per-wave LDS to re-enter A-frag layout
#pragma unroll
        for (int nt = 0; nt < 4; ++nt)
#pragma unroll
            for (int v = 0; v < 8; ++v) {
                const float p = __expf(sf[nt][v] - rmax[v]);
                lsum[v] += p;
                sP[wave][(hlf * 8 + v) * VSTR + nt * 16 + l16] = f2bf(p);
            }
#pragma unroll
        for (int off = 1; off < 16; off <<= 1)
#pragma unroll
            for (int v = 0; v < 8; ++v) lsum[v] += __shfl_xor(lsum[v], off, 32);
#pragma unroll
        for (int v = 0; v < 8; ++v) rsum[v] += lsum[v];
        // out += P(16x64) * V(64x128); batch V-fragments in groups of 4
#pragma unroll
        for (int kc2 = 0; kc2 < 2; ++kc2) {
            Frag16 pa;
            const int prow = l16 * VSTR + kc2 * 32;
            pa.u[0] = *(const uint4*)&sP[wave][prow + hlf * 8];
            pa.u[1] = *(const uint4*)&sP[wave][prow + 16 + hlf * 8];
#pragma unroll
            for (int dg = 0; dg < 2; ++dg) {
                Frag16 vb[4];
#pragma unroll
                for (int j = 0; j < 4; ++j) {
                    const int vrow = ((dg * 4 + j) * 16 + l16) * VSTR + kc2 * 32 + hlf * 8;
                    vb[j].u[0] = *(const uint4*)&sVt[vrow];
                    vb[j].u[1] = *(const uint4*)&sVt[vrow + 16];
                }
#pragma unroll
                for (int j = 0; j < 4; ++j)
                    oacc[dg * 4 + j] = __builtin_amdgcn_wmma_f32_16x16x32_bf16(
                        false, pa.v, false, vb[j].v, (short)0, oacc[dg * 4 + j], false, false);
            }
        }
    }
    // epilogue: normalize, write [B, S, H*128] bf16 for the O-projection GEMM
#pragma unroll
    for (int dt = 0; dt < 8; ++dt)
#pragma unroll
        for (int v = 0; v < 8; ++v) {
            const int r = qbase + wave * 16 + hlf * 8 + v;
            const int c = h * HD + dt * 16 + l16;
            O[((size_t)b * SEQ + r) * DIM + c] = f2bf(oacc[dt][v] / rsum[v]);
        }
}

// ---------------------------------------------------------------- launch ----
extern "C" void kernel_launch(void* const* d_in, const int* in_sizes, int n_in,
                              void* d_out, int out_size, void* d_ws, size_t ws_size,
                              hipStream_t stream) {
    const float* hidden = (const float*)d_in[0];
    const float* cosb   = (const float*)d_in[2];
    const float* sinb   = (const float*)d_in[3];
    const float* Wq     = (const float*)d_in[5];
    const float* Wk     = (const float*)d_in[6];
    const float* Wv     = (const float*)d_in[7];
    const float* Wo     = (const float*)d_in[8];
    float* out = (float*)d_out;

    char* ws = (char*)d_ws;
    size_t off = 0;
    auto alloc = [&](size_t bytes) -> void* {
        void* p = ws + off;
        off += (bytes + 255) & ~(size_t)255;
        return p;
    };
    const size_t M = (size_t)BATCH * SEQ;                 // 2048
    unsigned short* Xbf = (unsigned short*)alloc(M * DIM * 2);
    unsigned short* WqT = (unsigned short*)alloc((size_t)DIM * DIM * 2);
    unsigned short* WkT = (unsigned short*)alloc((size_t)(NKV * HD) * DIM * 2);
    unsigned short* WvT = (unsigned short*)alloc((size_t)(NKV * HD) * DIM * 2);
    unsigned short* WoT = (unsigned short*)alloc((size_t)DIM * DIM * 2);
    unsigned short* Qp  = (unsigned short*)alloc(M * DIM * 2);
    unsigned short* Kp  = (unsigned short*)alloc(M * NKV * HD * 2);
    unsigned short* Vp  = (unsigned short*)alloc(M * NKV * HD * 2);
    unsigned short* Qr  = (unsigned short*)alloc(M * DIM * 2);
    unsigned short* Krp = (unsigned short*)alloc(M * NKV * HD * 2);
    unsigned short* Vr  = (unsigned short*)alloc(M * NKV * HD * 2);
    unsigned short* Att = (unsigned short*)alloc(M * DIM * 2);

    const long nh = (long)M * DIM;
    f32_to_bf16_kernel<<<(nh + 255) / 256, 256, 0, stream>>>(hidden, Xbf, nh);

    dim3 tb(32, 8);
    transpose_to_bf16_kernel<<<dim3(DIM / 32, DIM / 32), tb, 0, stream>>>(Wq, WqT, DIM, DIM);
    transpose_to_bf16_kernel<<<dim3((NKV * HD) / 32, DIM / 32), tb, 0, stream>>>(Wk, WkT, DIM, NKV * HD);
    transpose_to_bf16_kernel<<<dim3((NKV * HD) / 32, DIM / 32), tb, 0, stream>>>(Wv, WvT, DIM, NKV * HD);
    transpose_to_bf16_kernel<<<dim3(DIM / 32, DIM / 32), tb, 0, stream>>>(Wo, WoT, DIM, DIM);

    gemm_bf16_kernel<1><<<dim3(DIM / GBN, M / GBM), 256, 0, stream>>>(Xbf, WqT, Qp, M, DIM, DIM);
    gemm_bf16_kernel<1><<<dim3((NKV * HD) / GBN, M / GBM), 256, 0, stream>>>(Xbf, WkT, Kp, M, NKV * HD, DIM);
    gemm_bf16_kernel<1><<<dim3((NKV * HD) / GBN, M / GBM), 256, 0, stream>>>(Xbf, WvT, Vp, M, NKV * HD, DIM);

    const long nq = (long)BATCH * SEQ * NH * 64;
    rope_permute_kernel<<<(nq + 255) / 256, 256, 0, stream>>>(Qp, cosb, sinb, Qr, NH, nq);
    const long nk = (long)BATCH * SEQ * NKV * 64;
    rope_permute_kernel<<<(nk + 255) / 256, 256, 0, stream>>>(Kp, cosb, sinb, Krp, NKV, nk);
    const long nv = (long)BATCH * SEQ * NKV * HD;
    permute_heads_kernel<<<(nv + 255) / 256, 256, 0, stream>>>(Vp, Vr, NKV, nv);

    attention_kernel<<<dim3(SEQ / 64, NH, BATCH), 128, 0, stream>>>(Qr, Krp, Vr, Att);

    gemm_bf16_kernel<0><<<dim3(DIM / GBN, M / GBM), 256, 0, stream>>>(Att, WoT, out, M, DIM, DIM);
    (void)in_sizes; (void)n_in; (void)out_size; (void)ws_size;
}
